// FactoredConv2d_6210522710118
// MI455X (gfx1250) — matrix operand order repaired and checked by reading, not compiled, not run on hardware
//
#include <hip/hip_runtime.h>

// ---------------------------------------------------------------------------
// DoRA+LoRA fused linear:  y = x @ W_total^T   with
//   W_total = Q * ((scale_A * g) @ scale_B) + lora_B @ lora_A   (scaling = 1.0)
// bf16x3 split-precision WMMA GEMM, double-buffered LDS fed by CDNA5
// GLOBAL_LOAD_ASYNC_TO_LDS_B128 (ASYNCcnt-tracked, no VGPR staging).
// Steady-state K-loop is branch-free (last iteration peeled).
// ---------------------------------------------------------------------------

typedef __attribute__((ext_vector_type(16))) __bf16 v16bf;
typedef __attribute__((ext_vector_type(8)))  float  v8f;
typedef __attribute__((ext_vector_type(4)))  unsigned short ushort4v;

#define M_DIM 4096   // B*S
#define N_DIM 4096   // OUT
#define K_DIM 4096   // IN
#define RANK_ 8
#define LORA_R_ 16
#define LORA_SCALING 1.0f

#define BM 128
#define BN 128
#define BK 32
#define LDSK 40      // padded LDS row stride in halves (80B, 16B-aligned rows)

__device__ __forceinline__ unsigned short bf16_rne(float f) {
  unsigned int u = __float_as_uint(f);
  u += 0x7FFFu + ((u >> 16) & 1u);   // round-to-nearest-even
  return (unsigned short)(u >> 16);
}
__device__ __forceinline__ float bf16_to_f32(unsigned short h) {
  return __uint_as_float(((unsigned int)h) << 16);
}
__device__ __forceinline__ void split_bf16(float w, unsigned short& hi, unsigned short& lo) {
  hi = bf16_rne(w);
  lo = bf16_rne(w - bf16_to_f32(hi));
}

// Async global->LDS copy, GVS addressing: 16B per lane, tracked by ASYNCcnt.
#define ASYNC_CP(ldsoff, goff, base)                                   \
  asm volatile("global_load_async_to_lds_b128 %0, %1, %2"              \
               :: "v"(ldsoff), "v"(goff), "s"(base) : "memory")

#if __has_builtin(__builtin_amdgcn_s_wait_asynccnt)
#define WAIT_ASYNC(n) __builtin_amdgcn_s_wait_asynccnt(n)
#else
#define WAIT_ASYNC(n) asm volatile("s_wait_asynccnt %0" :: "i"(n) : "memory")
#endif

// ---------------------------------------------------------------------------
// Kernel 1: W_total = Q * S + lora ; split to bf16 hi/lo.  One block per row o.
// ---------------------------------------------------------------------------
__global__ __launch_bounds__(256) void dora_prep_w(
    const float* __restrict__ weight, const float* __restrict__ sA,
    const float* __restrict__ sB, const float* __restrict__ g,
    const float* __restrict__ lA, const float* __restrict__ lB,
    unsigned short* __restrict__ Whi, unsigned short* __restrict__ Wlo) {
  const int o = blockIdx.x;
  float a[RANK_], b[LORA_R_];
#pragma unroll
  for (int k = 0; k < RANK_; ++k) a[k] = sA[o * RANK_ + k] * g[k];
#pragma unroll
  for (int r = 0; r < LORA_R_; ++r) b[r] = lB[o * LORA_R_ + r];
  const long rowBase = (long)o * K_DIM;
  for (int i = threadIdx.x; i < K_DIM; i += 256) {
    float s = 0.f, l = 0.f;
#pragma unroll
    for (int k = 0; k < RANK_; ++k) s += a[k] * sB[k * K_DIM + i];
#pragma unroll
    for (int r = 0; r < LORA_R_; ++r) l += b[r] * lA[r * K_DIM + i];
    float w = weight[rowBase + i] * s + LORA_SCALING * l;
    unsigned short hi, lo;
    split_bf16(w, hi, lo);
    Whi[rowBase + i] = hi;
    Wlo[rowBase + i] = lo;
  }
}

// ---------------------------------------------------------------------------
// Kernel 2: split x (fp32) into bf16 hi/lo arrays. 4 elems / thread (float4).
// ---------------------------------------------------------------------------
__global__ __launch_bounds__(256) void dora_prep_x(
    const float4* __restrict__ x, ushort4v* __restrict__ xhi,
    ushort4v* __restrict__ xlo) {
  const long i = (long)blockIdx.x * 256 + threadIdx.x;
  float4 v = x[i];
  ushort4v h, l;
  split_bf16(v.x, ((unsigned short*)&h)[0], ((unsigned short*)&l)[0]);
  split_bf16(v.y, ((unsigned short*)&h)[1], ((unsigned short*)&l)[1]);
  split_bf16(v.z, ((unsigned short*)&h)[2], ((unsigned short*)&l)[2]);
  split_bf16(v.w, ((unsigned short*)&h)[3], ((unsigned short*)&l)[3]);
  xhi[i] = h;
  xlo[i] = l;
}

// ---------------------------------------------------------------------------
// Kernel 3: bf16x3 WMMA GEMM.  y[m,n] = sum_k x[m,k] * W[n,k]
// 256 threads = 8 waves. Tile 128x128, BK=32, double-buffered async LDS.
// Wave grid 4(M) x 2(N): each wave -> 32x64 patch = 2x4 subtiles of 16x16.
// ---------------------------------------------------------------------------
union Frag {
  int4 q[2];
  v16bf v;
};

__global__ __launch_bounds__(256) void dora_gemm(
    const unsigned short* __restrict__ Xhi, const unsigned short* __restrict__ Xlo,
    const unsigned short* __restrict__ Whi, const unsigned short* __restrict__ Wlo,
    float* __restrict__ out) {
  __shared__ __align__(16) unsigned short Ahi[2][BM * LDSK];
  __shared__ __align__(16) unsigned short Alo[2][BM * LDSK];
  __shared__ __align__(16) unsigned short Bhi[2][BN * LDSK];
  __shared__ __align__(16) unsigned short Blo[2][BN * LDSK];

  const int tid   = threadIdx.x;
  const int lane  = tid & 31;
  const int wave  = tid >> 5;
  const int mOff  = (wave >> 1) * 32;   // 0,32,64,96
  const int nOff  = (wave & 1) * 64;    // 0,64
  const int mBase = blockIdx.y * BM;
  const int nBase = blockIdx.x * BN;

  const int r16    = lane & 15;
  const int akb    = (lane < 16) ? 0 : 8;    // A frag K base (lane-half pattern)
  const int bkb    = (lane < 16) ? 0 : 8;    // B frag K base
  const int rowAdd = (lane < 16) ? 0 : 8;    // C/D row offset for upper lanes

  v8f acc[2][4] = {};

  // global->LDS staging map: 256 threads cover 64 rows x 32 cols (8-half chunks)
  const int ldRow = tid >> 2;        // 0..63
  const int ldCol = (tid & 3) * 8;   // 0,8,16,24

  // LDS byte offsets for async destinations (low 32 bits of generic shared ptr
  // are the wave-relative LDS byte address).
  const unsigned ldsAhi = (unsigned)(unsigned long long)&Ahi[0][0];
  const unsigned ldsAlo = (unsigned)(unsigned long long)&Alo[0][0];
  const unsigned ldsBhi = (unsigned)(unsigned long long)&Bhi[0][0];
  const unsigned ldsBlo = (unsigned)(unsigned long long)&Blo[0][0];
  const unsigned bufBytes = BM * LDSK * 2u;          // one buffer stride
  const unsigned stOff    = (unsigned)((ldRow * LDSK + ldCol) * 2);
  const unsigned stRow1   = 64u * LDSK * 2u;         // +64 rows in LDS
  const unsigned gRow1    = 64u * K_DIM * 2u;        // +64 rows in global
  const unsigned gA0 = (unsigned)(((long)(mBase + ldRow) * K_DIM + ldCol) * 2);
  const unsigned gB0 = (unsigned)(((long)(nBase + ldRow) * K_DIM + ldCol) * 2);

  auto stage = [&](int buf, unsigned kByte) {
    const unsigned l0 = stOff + (unsigned)buf * bufBytes;
    const unsigned a0 = gA0 + kByte;
    const unsigned b0 = gB0 + kByte;
    ASYNC_CP(ldsAhi + l0, a0, Xhi);
    ASYNC_CP(ldsAlo + l0, a0, Xlo);
    ASYNC_CP(ldsBhi + l0, b0, Whi);
    ASYNC_CP(ldsBlo + l0, b0, Wlo);
    ASYNC_CP(ldsAhi + l0 + stRow1, a0 + gRow1, Xhi);
    ASYNC_CP(ldsAlo + l0 + stRow1, a0 + gRow1, Xlo);
    ASYNC_CP(ldsBhi + l0 + stRow1, b0 + gRow1, Whi);
    ASYNC_CP(ldsBlo + l0 + stRow1, b0 + gRow1, Wlo);
  };

  auto compute = [&](int buf) {
    // A fragments (16x32 bf16): per lane K = {akb..akb+7, akb+16..akb+23}
    Frag aHi[2], aLo[2];
#pragma unroll
    for (int ms = 0; ms < 2; ++ms) {
      const int base = (mOff + ms * 16 + r16) * LDSK;
      aHi[ms].q[0] = *(const int4*)&Ahi[buf][base + akb];
      aHi[ms].q[1] = *(const int4*)&Ahi[buf][base + akb + 16];
      aLo[ms].q[0] = *(const int4*)&Alo[buf][base + akb];
      aLo[ms].q[1] = *(const int4*)&Alo[buf][base + akb + 16];
    }
#pragma unroll
    for (int ns = 0; ns < 4; ++ns) {
      // B fragments (32x16 bf16): per lane (col = r16) K = {bkb..bkb+7, bkb+16..bkb+23}
      Frag bHi, bLo;
      const int base = (nOff + ns * 16 + r16) * LDSK;
      bHi.q[0] = *(const int4*)&Bhi[buf][base + bkb];
      bHi.q[1] = *(const int4*)&Bhi[buf][base + bkb + 16];
      bLo.q[0] = *(const int4*)&Blo[buf][base + bkb];
      bLo.q[1] = *(const int4*)&Blo[buf][base + bkb + 16];
#pragma unroll
      for (int ms = 0; ms < 2; ++ms) {
        acc[ms][ns] = __builtin_amdgcn_wmma_f32_16x16x32_bf16(
            false, aHi[ms].v, false, bHi.v, (short)0, acc[ms][ns], false, false);
        acc[ms][ns] = __builtin_amdgcn_wmma_f32_16x16x32_bf16(
            false, aHi[ms].v, false, bLo.v, (short)0, acc[ms][ns], false, false);
        acc[ms][ns] = __builtin_amdgcn_wmma_f32_16x16x32_bf16(
            false, aLo[ms].v, false, bHi.v, (short)0, acc[ms][ns], false, false);
      }
    }
  };

  stage(0, 0);                         // prefetch tile 0
  int buf = 0;

  // Steady state: branch-free — prefetch next, wait current, compute current.
  for (int k0 = 0; k0 < K_DIM - BK; k0 += BK) {
    stage(buf ^ 1, (unsigned)((k0 + BK) * 2));
    WAIT_ASYNC(8);     // <=8 outstanding (in-order) => current tile landed
    __syncthreads();   // current tile visible to all waves
    compute(buf);
    __syncthreads();   // readers done before buffer is overwritten next iter
    buf ^= 1;
  }

  // Peeled final K-step: no prefetch, drain ASYNCcnt fully.
  WAIT_ASYNC(0);
  __syncthreads();
  compute(buf);

  // Epilogue: C/D layout — VGPR r: lanes 0-15 -> row r, lanes 16-31 -> row r+8
#pragma unroll
  for (int ms = 0; ms < 2; ++ms) {
#pragma unroll
    for (int ns = 0; ns < 4; ++ns) {
      const int col = nBase + nOff + ns * 16 + r16;
#pragma unroll
      for (int r = 0; r < 8; ++r) {
        const int row = mBase + mOff + ms * 16 + r + rowAdd;
        out[(long)row * N_DIM + col] = acc[ms][ns][r];
      }
    }
  }
}

// ---------------------------------------------------------------------------
extern "C" void kernel_launch(void* const* d_in, const int* in_sizes, int n_in,
                              void* d_out, int out_size, void* d_ws, size_t ws_size,
                              hipStream_t stream) {
  const float* x      = (const float*)d_in[0];
  const float* weight = (const float*)d_in[1];
  const float* sA     = (const float*)d_in[2];
  const float* sB     = (const float*)d_in[3];
  const float* g      = (const float*)d_in[4];
  const float* lA     = (const float*)d_in[5];
  const float* lB     = (const float*)d_in[6];
  float* out          = (float*)d_out;

  // Workspace: Whi | Wlo | Xhi | Xlo  (bf16 each, 32MB each = 128MB total)
  unsigned short* Whi = (unsigned short*)d_ws;
  unsigned short* Wlo = Whi + (size_t)N_DIM * K_DIM;
  unsigned short* Xhi = Wlo + (size_t)N_DIM * K_DIM;
  unsigned short* Xlo = Xhi + (size_t)M_DIM * K_DIM;

  dora_prep_w<<<N_DIM, 256, 0, stream>>>(weight, sA, sB, g, lA, lB, Whi, Wlo);

  const long xquads = (long)M_DIM * K_DIM / 4;           // 4 floats per thread
  dora_prep_x<<<(unsigned)(xquads / 256), 256, 0, stream>>>(
      (const float4*)x, (ushort4v*)Xhi, (ushort4v*)Xlo);

  dim3 grid(N_DIM / BN, M_DIM / BM);
  dora_gemm<<<grid, 256, 0, stream>>>(Xhi, Xlo, Whi, Wlo, out);
}
